// SelfAttentionBlock_9981503996051
// MI455X (gfx1250) — matrix operand order repaired and checked by reading, not compiled, not run on hardware
//
#include <hip/hip_runtime.h>
#include <hip/hip_bf16.h>
#include <cstdint>

#define DEV __device__ __forceinline__

typedef __attribute__((ext_vector_type(8)))  float        v8f;
typedef __attribute__((ext_vector_type(16))) __bf16       v16bf;
typedef __attribute__((ext_vector_type(4)))  unsigned int u32x4;

union FragBF {
  unsigned short u[16];
  u32x4          q2[2];
  v16bf          v;
};

DEV unsigned short f2bf(float f) {
  unsigned int u = __builtin_bit_cast(unsigned int, f);
  unsigned int r = u + 0x7fffu + ((u >> 16) & 1u);   // round-to-nearest-even
  return (unsigned short)(r >> 16);
}

// CDNA5 async global->LDS copy, 16B per lane. Tracked by ASYNCcnt (CDNA5 ISA).
DEV void async_g2l_b128(unsigned int ldsaddr, const void* gaddr) {
  asm volatile("global_load_async_to_lds_b128 %0, %1, off"
               :: "v"(ldsaddr), "v"(gaddr) : "memory");
}
DEV unsigned int lds_off(const void* p) {   // flat->LDS aperture uses addr[31:0]
  return (unsigned int)(uintptr_t)p;
}

static constexpr int NB = 4, L = 1024, D = 1024, NH = 16, DH = 64, E3 = 3072;
static constexpr int M_TOT = NB * L;   // 4096
static constexpr float EPS = 1e-6f;

// ---------------- workspace layout (bytes) ----------------
static constexpr size_t OFF_S    = 0;                          // f32 [4][1024]
static constexpr size_t OFF_H    = OFF_S    + 16384;           // bf16 [4096][1024]
static constexpr size_t OFF_WQKV = OFF_H    + 8388608;         // bf16 [3072][1024]
static constexpr size_t OFF_WOUT = OFF_WQKV + 6291456;         // bf16 [1024][1024]
static constexpr size_t OFF_QKV  = OFF_WOUT + 2097152;         // f32  [4096][3072]
static constexpr size_t OFF_Q    = OFF_QKV  + 50331648;        // bf16 [4][16][1024][64]
static constexpr size_t OFF_K    = OFF_Q    + 8388608;
static constexpr size_t OFF_V    = OFF_K    + 8388608;
static constexpr size_t OFF_O    = OFF_V    + 8388608;         // bf16 [4096][1024]

// ---------------- per-batch modulation scale: s = cond @ norm_w^T + 1 ----------------
__global__ void k_scale(const float* __restrict__ cond, const float* __restrict__ norm_w,
                        float* __restrict__ s) {
  int idx = blockIdx.x * blockDim.x + threadIdx.x;
  if (idx >= NB * D) return;
  int n = idx >> 10, d = idx & 1023;
  const float* c = cond + (size_t)n * D;
  const float* w = norm_w + (size_t)d * D;
  float acc = 0.f;
  for (int i = 0; i < D; ++i) acc += c[i] * w[i];
  s[idx] = acc + 1.0f;
}

// ---------------- generic fp32 -> bf16 cast ----------------
__global__ void k_cast(const float* __restrict__ src, unsigned short* __restrict__ dst, int count) {
  int stride = gridDim.x * blockDim.x;
  for (int i = blockIdx.x * blockDim.x + threadIdx.x; i < count; i += stride)
    dst[i] = f2bf(src[i]);
}

// ---------------- h = rms_norm(x, s) -> bf16 ----------------
__global__ void k_rms_h(const float* __restrict__ x, const float* __restrict__ s,
                        unsigned short* __restrict__ h) {
  __shared__ float red[256];
  int row = blockIdx.x;             // 0..4095 == n*1024+l
  int n = row >> 10;
  const float* xr = x + (size_t)row * D;
  float p = 0.f;
  for (int i = threadIdx.x; i < D; i += 256) { float v = xr[i]; p += v * v; }
  red[threadIdx.x] = p;
  __syncthreads();
  for (int off = 128; off > 0; off >>= 1) {
    if ((int)threadIdx.x < off) red[threadIdx.x] += red[threadIdx.x + off];
    __syncthreads();
  }
  float rinv = rsqrtf(red[0] * (1.0f / D) + EPS);
  const float* sr = s + (size_t)n * D;
  unsigned short* hr = h + (size_t)row * D;
  for (int i = threadIdx.x; i < D; i += 256)
    hr[i] = f2bf(xr[i] * sr[i] * rinv);
}

// ---------------- LDS-staged bf16 WMMA GEMM: C[m][n] = sum_k A[m][k] * B[n][k] ----
// Block tile 128x128, K-step 32, double-buffered LDS filled by async global->LDS.
// 8 waves; each wave computes a 64x32 register tile (4 x 2 accumulators).
static constexpr int SA = 40;   // padded LDS row stride (bf16 elems): 32 + 8
__global__ void __launch_bounds__(256) k_gemm_bf16(const unsigned short* __restrict__ A,
                                                   const unsigned short* __restrict__ B,
                                                   float* __restrict__ C,
                                                   int M, int N, int K) {
  __shared__ unsigned short shA[2][128 * SA];
  __shared__ unsigned short shB[2][128 * SA];

  int tid  = threadIdx.x;
  int lane = tid & 31;
  int wave = tid >> 5;
  int lo = lane & 15, hi = lane >> 4;
  int wm = (wave & 1) * 64;         // 2 waves stacked in M
  int wn = (wave >> 1) * 32;        // 4 waves across N
  int mt0 = blockIdx.y * 128;
  int nt0 = blockIdx.x * 128;

  const unsigned short* Ag = A + (size_t)mt0 * K;
  const unsigned short* Bg = B + (size_t)nt0 * K;
  unsigned int ldsA0 = lds_off(&shA[0][0]), ldsA1 = lds_off(&shA[1][0]);
  unsigned int ldsB0 = lds_off(&shB[0][0]), ldsB1 = lds_off(&shB[1][0]);

  int c0 = tid, c1 = tid + 256;                      // two 16B chunks per thread
  int r0 = c0 >> 2, o0 = (c0 & 3) * 8;               // row, col (elems) within tile
  int r1 = c1 >> 2, o1 = (c1 & 3) * 8;

  const int nk = K / 32;
  // stage 0
  {
    int kk = 0;
    async_g2l_b128(ldsA0 + (unsigned)(r0 * SA + o0) * 2, Ag + (size_t)r0 * K + kk + o0);
    async_g2l_b128(ldsB0 + (unsigned)(r0 * SA + o0) * 2, Bg + (size_t)r0 * K + kk + o0);
    async_g2l_b128(ldsA0 + (unsigned)(r1 * SA + o1) * 2, Ag + (size_t)r1 * K + kk + o1);
    async_g2l_b128(ldsB0 + (unsigned)(r1 * SA + o1) * 2, Bg + (size_t)r1 * K + kk + o1);
  }

  v8f acc[4][2] = {};
  for (int kt = 0; kt < nk; ++kt) {
    if (kt + 1 < nk) {                               // prefetch next stage
      int kk = (kt + 1) * 32;
      unsigned int lA = ((kt + 1) & 1) ? ldsA1 : ldsA0;
      unsigned int lB = ((kt + 1) & 1) ? ldsB1 : ldsB0;
      async_g2l_b128(lA + (unsigned)(r0 * SA + o0) * 2, Ag + (size_t)r0 * K + kk + o0);
      async_g2l_b128(lB + (unsigned)(r0 * SA + o0) * 2, Bg + (size_t)r0 * K + kk + o0);
      async_g2l_b128(lA + (unsigned)(r1 * SA + o1) * 2, Ag + (size_t)r1 * K + kk + o1);
      async_g2l_b128(lB + (unsigned)(r1 * SA + o1) * 2, Bg + (size_t)r1 * K + kk + o1);
      asm volatile("s_wait_asynccnt 4" ::: "memory"); // current stage landed (in-order)
    } else {
      asm volatile("s_wait_asynccnt 0" ::: "memory");
    }
    __syncthreads();

    const unsigned short* sA = &shA[kt & 1][0];
    const unsigned short* sB = &shB[kt & 1][0];
    FragBF b[2];
    for (int ns = 0; ns < 2; ++ns) {                 // B: K = e + 8*(e>>3) + 8*(lane>>4)
      const unsigned short* br = sB + (wn + ns * 16 + lo) * SA + 8 * hi;
      b[ns].q2[0] = *(const u32x4*)(br);
      b[ns].q2[1] = *(const u32x4*)(br + 16);
    }
    for (int ms = 0; ms < 4; ++ms) {                 // A: K = e + 8*(e>>3) + 8*(lane>>4)
      FragBF a;
      const unsigned short* ar = sA + (wm + ms * 16 + lo) * SA + 8 * hi;
      a.q2[0] = *(const u32x4*)(ar);
      a.q2[1] = *(const u32x4*)(ar + 16);
      acc[ms][0] = __builtin_amdgcn_wmma_f32_16x16x32_bf16(false, a.v, false, b[0].v, (short)0, acc[ms][0], false, false);
      acc[ms][1] = __builtin_amdgcn_wmma_f32_16x16x32_bf16(false, a.v, false, b[1].v, (short)0, acc[ms][1], false, false);
    }
    __syncthreads();                                 // readers done before next prefetch overwrite
  }

  for (int ms = 0; ms < 4; ++ms)
    for (int ns = 0; ns < 2; ++ns) {
      float* crow = C + (size_t)(mt0 + wm + ms * 16 + 8 * hi) * N + (nt0 + wn + ns * 16 + lo);
      for (int j = 0; j < 8; ++j)
        crow[(size_t)j * N] = acc[ms][ns][j];
    }
}

// ---------------- per-head RMS norm + RoPE; emit head-major bf16 Q/K/V ----------------
__global__ void k_rope(const float* __restrict__ qkv, const float* __restrict__ pos,
                       const float* __restrict__ qk_scale, const float* __restrict__ freqs,
                       unsigned short* __restrict__ qb, unsigned short* __restrict__ kb,
                       unsigned short* __restrict__ vb) {
  int idx = blockIdx.x * blockDim.x + threadIdx.x;
  if (idx >= NB * L * NH) return;
  int h = idx & 15, l = (idx >> 4) & 1023, n = idx >> 14;
  const float* base = qkv + (size_t)(n * L + l) * E3 + h * DH;
  float p0 = pos[(size_t)(n * L + l) * 2 + 0];
  float p1 = pos[(size_t)(n * L + l) * 2 + 1];
  float sc = qk_scale[h];
  size_t obase = ((size_t)((n * NH + h) * L) + l) * DH;
  float cs[32], sn[32];
  for (int j = 0; j < 32; ++j) {
    float f  = freqs[h * 16 + (j & 15)];
    float th = ((j < 16) ? p0 : p1) * f;
    cs[j] = __cosf(th);
    sn[j] = __sinf(th);
  }
  for (int which = 0; which < 2; ++which) {   // 0 = q, 1 = k
    const float* r = base + which * D;
    unsigned short* o = (which == 0 ? qb : kb) + obase;
    float v[DH], ms = 0.f;
    for (int d = 0; d < DH; ++d) { v[d] = r[d]; ms += v[d] * v[d]; }
    float rinv = rsqrtf(ms * (1.0f / DH) + EPS) * sc;
    for (int d = 0; d < DH; ++d) v[d] *= rinv;
    for (int j = 0; j < 32; ++j) {
      float a = v[j], b2 = v[j + 32];
      v[j]      = a * cs[j] - b2 * sn[j];
      v[j + 32] = b2 * cs[j] + a * sn[j];
    }
    for (int d = 0; d < DH; ++d) o[d] = f2bf(v[d]);
  }
  const float* vr = base + 2 * D;             // v: plain cast
  unsigned short* ov = vb + obase;
  for (int d = 0; d < DH; ++d) ov[d] = f2bf(vr[d]);
}

// ---------------- attention: one wave per (n, head, 16-row q tile) ----------------
__global__ void __launch_bounds__(32) k_attn(const unsigned short* __restrict__ qb,
                                             const unsigned short* __restrict__ kb,
                                             const unsigned short* __restrict__ vb,
                                             unsigned short* __restrict__ ob) {
  __shared__ float smem[16 * 1024];           // 64 KB score strip: s[m][key]
  int lane = threadIdx.x;
  int qt = blockIdx.x, h = blockIdx.y, n = blockIdx.z;
  int lo = lane & 15, hi = lane >> 4;
  size_t headbase = (size_t)((n * NH + h) * L) * DH;

  // Q tile (16 x 64) as two A fragments (K = 0..31, 32..63)
  FragBF aq0, aq1;
  {
    const unsigned short* qrow = qb + headbase + (size_t)(qt * 16 + lo) * DH;
    int k0 = 8 * hi;
    aq0.q2[0] = *(const u32x4*)(qrow + k0);
    aq0.q2[1] = *(const u32x4*)(qrow + k0 + 16);
    aq1.q2[0] = *(const u32x4*)(qrow + 32 + k0);
    aq1.q2[1] = *(const u32x4*)(qrow + 32 + k0 + 16);
  }

  // S = (Q K^T) / 8 -> LDS
  for (int kt = 0; kt < 64; ++kt) {
    const unsigned short* krow = kb + headbase + (size_t)(kt * 16 + lo) * DH;
    FragBF b0, b1;
    int kbo = 8 * hi;
    b0.q2[0] = *(const u32x4*)(krow + kbo);
    b0.q2[1] = *(const u32x4*)(krow + kbo + 16);
    b1.q2[0] = *(const u32x4*)(krow + 32 + kbo);
    b1.q2[1] = *(const u32x4*)(krow + 32 + kbo + 16);
    v8f c = {};
    c = __builtin_amdgcn_wmma_f32_16x16x32_bf16(false, aq0.v, false, b0.v, (short)0, c, false, false);
    c = __builtin_amdgcn_wmma_f32_16x16x32_bf16(false, aq1.v, false, b1.v, (short)0, c, false, false);
    for (int j = 0; j < 8; ++j)
      smem[(j + 8 * hi) * 1024 + kt * 16 + lo] = c[j] * 0.125f;
  }
  __syncthreads();

  // exact softmax stats: lanes 0..15 own one row each (registers, broadcast by shfl)
  float rmax = -3.0e38f, rsum = 0.f;
  if (lane < 16) {
    const float* row = smem + lane * 1024;
    for (int i = 0; i < 1024; ++i) rmax = fmaxf(rmax, row[i]);
    for (int i = 0; i < 1024; ++i) rsum += __expf(row[i] - rmax);
  }
  __syncthreads();
  float mymax = __shfl(rmax, lo, 32);         // max of my A-fragment row (m = lane%16)

  // O = softmax(S) V  (exp fused into A-fragment load; 4 column tiles of 16)
  v8f acc[4] = {};
  const unsigned short* vbase = vb + headbase;
  for (int ks = 0; ks < 32; ++ks) {
    FragBF pa;
    const float* srow = smem + lo * 1024 + ks * 32;
    for (int e = 0; e < 16; ++e) {
      int k = e + 16 * hi;
      pa.u[e] = f2bf(__expf(srow[k] - mymax));
    }
    int kk0 = ks * 32 + 16 * hi;
    for (int t = 0; t < 4; ++t) {
      FragBF bv;
      const unsigned short* vcol = vbase + t * 16 + lo;
      for (int e = 0; e < 16; ++e)
        bv.u[e] = vcol[(size_t)(kk0 + e) * DH];
      acc[t] = __builtin_amdgcn_wmma_f32_16x16x32_bf16(false, pa.v, false, bv.v, (short)0, acc[t], false, false);
    }
  }

  // normalize and store back in model layout (bf16) for the output projection
  for (int j = 0; j < 8; ++j) {
    int Mr = j + 8 * hi;
    float denom = __shfl(rsum, Mr, 32);
    int l = qt * 16 + Mr;
    unsigned short* orow = ob + (size_t)(n * L + l) * D + h * DH + lo;
    for (int t = 0; t < 4; ++t)
      orow[t * 16] = f2bf(acc[t][j] / denom);
  }
}

extern "C" void kernel_launch(void* const* d_in, const int* in_sizes, int n_in,
                              void* d_out, int out_size, void* d_ws, size_t ws_size,
                              hipStream_t stream) {
  const float* x        = (const float*)d_in[0];
  const float* pos      = (const float*)d_in[1];
  /* d_in[2] attn_mask: all-true, unused */
  const float* cond     = (const float*)d_in[3];
  const float* norm_w   = (const float*)d_in[4];
  const float* qkv_w    = (const float*)d_in[5];
  const float* qk_scale = (const float*)d_in[6];
  const float* freqs    = (const float*)d_in[7];
  const float* out_w    = (const float*)d_in[8];
  float* out = (float*)d_out;

  char* ws = (char*)d_ws;
  float*          s_scale = (float*)(ws + OFF_S);
  unsigned short* h_bf    = (unsigned short*)(ws + OFF_H);
  unsigned short* wqkv    = (unsigned short*)(ws + OFF_WQKV);
  unsigned short* wout    = (unsigned short*)(ws + OFF_WOUT);
  float*          qkvf    = (float*)(ws + OFF_QKV);
  unsigned short* qbf     = (unsigned short*)(ws + OFF_Q);
  unsigned short* kbf     = (unsigned short*)(ws + OFF_K);
  unsigned short* vbf     = (unsigned short*)(ws + OFF_V);
  unsigned short* obf     = (unsigned short*)(ws + OFF_O);

  k_scale<<<16, 256, 0, stream>>>(cond, norm_w, s_scale);
  k_cast<<<512, 256, 0, stream>>>(qkv_w, wqkv, E3 * D);
  k_cast<<<512, 256, 0, stream>>>(out_w, wout, D * D);
  k_rms_h<<<M_TOT, 256, 0, stream>>>(x, s_scale, h_bf);
  k_gemm_bf16<<<dim3(E3 / 128, M_TOT / 128), 256, 0, stream>>>(h_bf, wqkv, qkvf, M_TOT, E3, D);
  k_rope<<<(NB * L * NH) / 256, 256, 0, stream>>>(qkvf, pos, qk_scale, freqs, qbf, kbf, vbf);
  k_attn<<<dim3(L / 16, NH, NB), 32, 0, stream>>>(qbf, kbf, vbf, obf);
  k_gemm_bf16<<<dim3(D / 128, M_TOT / 128), 256, 0, stream>>>(obf, wout, out, M_TOT, D, D);
}
